// MoELayer_48610439856256
// MI455X (gfx1250) — matrix operand order repaired and checked by reading, not compiled, not run on hardware
//
#include <hip/hip_runtime.h>
#include <hip/hip_bf16.h>

#define H    768
#define I    3072
#define NTOK 16384
#define E    8
#define ROWS 32
#define TILES_MAX (NTOK / ROWS) /* 512 */
#define XSTR 776                /* padded LDS row stride (halves) for x tile  */
#define HSTR 136                /* padded LDS row stride (halves) for h tile  */

typedef __bf16 bf16;
typedef __attribute__((ext_vector_type(16))) __bf16 v16bf;
typedef __attribute__((ext_vector_type(8)))  __bf16 v8bf;
typedef __attribute__((ext_vector_type(8)))  float  v8f;

union V16U { v16bf v; v8bf h[2]; };

__device__ __forceinline__ v8f v8f_zero() {
    v8f z = {0.f, 0.f, 0.f, 0.f, 0.f, 0.f, 0.f, 0.f};
    return z;
}

__device__ __forceinline__ float gelu_tanh(float v) {
    float c = 0.7978845608028654f * (v + 0.044715f * v * v * v);
    return 0.5f * v * (1.0f + tanhf(c));
}

// A-fragment (16x32 bf16) from LDS; p points at &lds[row*STR + kk + akoff].
// Layout: lanes 0-15 -> K {0..15}; lanes 16-31 -> K {16..31} (matches B).
__device__ __forceinline__ v16bf load_afrag(const bf16* p) {
    V16U a;
    a.h[0] = *(const v8bf*)p;
    a.h[1] = *(const v8bf*)(p + 8);
    return a.v;
}

// B-fragment (32x16): lane holds 16 consecutive K for its column.
template <bool BF16W>
__device__ __forceinline__ v16bf load_bfrag(const void* base, size_t elemOff) {
    V16U b;
    if constexpr (BF16W) {
        const bf16* p = (const bf16*)base + elemOff;
        b.h[0] = *(const v8bf*)p;
        b.h[1] = *(const v8bf*)(p + 8);
    } else {
        const float* p = (const float*)base + elemOff;
#pragma unroll
        for (int i = 0; i < 16; ++i) b.v[i] = (bf16)p[i];
    }
    return b.v;
}

// ---------------------------------------------------------------------------
// Kernel 0a: zero output + routing counters (graph-capture safe "memset")
// ---------------------------------------------------------------------------
__global__ void k_zero(float* __restrict__ y, int n,
                       int* __restrict__ counts, float* __restrict__ psum) {
    int tid = blockIdx.x * blockDim.x + threadIdx.x;
    for (int i = tid; i < n; i += gridDim.x * blockDim.x) y[i] = 0.f;
    if (tid < E) { counts[tid] = 0; psum[tid] = 0.f; }
}

// ---------------------------------------------------------------------------
// Kernel 0b: one-time f32 -> bf16 weight conversion into workspace
// ---------------------------------------------------------------------------
__global__ void k_cvtw(const float* __restrict__ src, bf16* __restrict__ dst, int n4) {
    int tid = blockIdx.x * blockDim.x + threadIdx.x;
    for (int i = tid; i < n4; i += gridDim.x * blockDim.x) {
        float4 f = *(const float4*)(src + (size_t)i * 4);
        bf16* d = dst + (size_t)i * 4;
        d[0] = (bf16)f.x; d[1] = (bf16)f.y; d[2] = (bf16)f.z; d[3] = (bf16)f.w;
    }
}

// ---------------------------------------------------------------------------
// Kernel 1: router — one wave32 per token.
// ---------------------------------------------------------------------------
__global__ void __launch_bounds__(256) k_router(
    const float* __restrict__ x, const float* __restrict__ gw,
    int* __restrict__ counts, float* __restrict__ psum,
    int* __restrict__ tok_idx, float* __restrict__ tok_w)
{
    __shared__ float g[E * H];                       // 24 KB
    for (int i = threadIdx.x; i < E * H; i += 256) g[i] = gw[i];
    __syncthreads();

    const int wave = threadIdx.x >> 5;
    const int lane = threadIdx.x & 31;
    const int n = blockIdx.x * 8 + wave;
    if (n >= NTOK) return;

    const float* xp = x + (size_t)n * H;
    float acc[E];
#pragma unroll
    for (int e = 0; e < E; ++e) acc[e] = 0.f;

    for (int j = lane; j < H; j += 32) {
        float xv = xp[j];
#pragma unroll
        for (int e = 0; e < E; ++e) acc[e] += xv * g[e * H + j];
    }
#pragma unroll
    for (int e = 0; e < E; ++e) {
#pragma unroll
        for (int off = 16; off > 0; off >>= 1)
            acc[e] += __shfl_xor(acc[e], off, 32);
    }

    if (lane == 0) {
        float mx = acc[0];
#pragma unroll
        for (int e = 1; e < E; ++e) mx = fmaxf(mx, acc[e]);
        float p[E], s = 0.f;
#pragma unroll
        for (int e = 0; e < E; ++e) { p[e] = __expf(acc[e] - mx); s += p[e]; }
        float inv = 1.f / s;
#pragma unroll
        for (int e = 0; e < E; ++e) { p[e] *= inv; atomicAdd(&psum[e], p[e]); }

        int i0 = 0;
#pragma unroll
        for (int e = 1; e < E; ++e) if (p[e] > p[i0]) i0 = e;
        int i1 = (i0 == 0) ? 1 : 0;
#pragma unroll
        for (int e = 0; e < E; ++e) if (e != i0 && p[e] > p[i1]) i1 = e;

        float w0 = p[i0], w1v = p[i1];
        float norm = 1.f / (w0 + w1v);
        w0 *= norm; w1v *= norm;

        int pos0 = atomicAdd(&counts[i0], 1);
        tok_idx[i0 * NTOK + pos0] = n;
        tok_w [i0 * NTOK + pos0] = w0;
        int pos1 = atomicAdd(&counts[i1], 1);
        tok_idx[i1 * NTOK + pos1] = n;
        tok_w [i1 * NTOK + pos1] = w1v;
    }
}

// ---------------------------------------------------------------------------
// Kernel 2: load-balancing aux loss -> d_out[N*H]
// ---------------------------------------------------------------------------
__global__ void k_aux(const int* __restrict__ counts,
                      const float* __restrict__ psum,
                      float* __restrict__ out_aux) {
    if (threadIdx.x == 0 && blockIdx.x == 0) {
        float s = 0.f;
        for (int e = 0; e < E; ++e) {
            float f = (float)counts[e] / (float)(NTOK * 2);
            float p = psum[e] / (float)NTOK;
            s += f * p;
        }
        *out_aux = 0.01f * (float)E * s;
    }
}

// ---------------------------------------------------------------------------
// Kernel 3: gathered expert FFN, bf16 WMMA, software-pipelined B-fragments.
// Block = (expert, tile of 32 gathered tokens), 8 waves.
// ---------------------------------------------------------------------------
template <bool BF16W>
__global__ void __launch_bounds__(256) k_ffn(
    const float* __restrict__ x,  const void* __restrict__ w1p,
    const void* __restrict__ w2p, const int* __restrict__ counts,
    const int* __restrict__ tok_idx, const float* __restrict__ tok_w,
    float* __restrict__ y)
{
    __shared__ bf16  x_lds[ROWS * XSTR];   // ~48.5 KB : 32 token rows (bf16)
    __shared__ bf16  h_lds[ROWS * HSTR];   // ~8.5 KB  : 32x128 activation chunk
    __shared__ int   trow[ROWS];
    __shared__ float wrow[ROWS];

    const int e    = blockIdx.x >> 9;             // blockIdx / TILES_MAX
    const int tile = blockIdx.x & (TILES_MAX - 1);
    const int cnt  = counts[e];
    const int row0 = tile * ROWS;
    if (row0 >= cnt) return;

    const int tid  = threadIdx.x;
    const int wave = tid >> 5;
    const int lane = tid & 31;
    const int ln16 = lane & 15;
    const int hi   = lane >> 4;

    if (tid < ROWS) {
        int gi = row0 + tid;
        trow[tid] = (gi < cnt) ? tok_idx[e * NTOK + gi] : -1;
        wrow[tid] = (gi < cnt) ? tok_w [e * NTOK + gi] : 0.f;
    }
    __syncthreads();

    // stage 32 token rows into LDS as bf16 (float4 loads, zero-pad short tiles)
    for (int idx = tid; idx < ROWS * (H / 4); idx += 256) {
        int r = idx / (H / 4), c4 = idx - r * (H / 4);
        int j = c4 * 4;
        int t = trow[r];
        bf16* d = &x_lds[r * XSTR + j];
        if (t >= 0) {
            float4 f = *(const float4*)&x[(size_t)t * H + j];
            d[0] = (bf16)f.x; d[1] = (bf16)f.y; d[2] = (bf16)f.z; d[3] = (bf16)f.w;
        } else {
            d[0] = (bf16)0.f; d[1] = (bf16)0.f; d[2] = (bf16)0.f; d[3] = (bf16)0.f;
        }
    }
    __syncthreads();

    const size_t wbytes = BF16W ? 2 : 4;
    const char* w1e = (const char*)w1p + (size_t)e * I * H * wbytes;
    const char* w2e = (const char*)w2p + (size_t)e * H * I * wbytes;

    v8f accY[12];
#pragma unroll
    for (int t = 0; t < 12; ++t) accY[t] = v8f_zero();

    const int akoff = hi ? 16 : 0;  // A-frag K offset (contiguous, matches B)
    const int bkoff = hi ? 16 : 0;  // B-frag K offset    (contiguous, matches A)

    for (int it = 0; it < I / 128; ++it) {
        const int ic0 = it * 128;

        // ---- GEMM1: this wave's 16 h-columns x 2 M-tiles ----
        // Fully unrolled with explicit 2-deep B double-buffer: next K-step's
        // B-fragment load is issued before the current step's WMMAs.
        const int ccol = wave * 16 + ln16;                    // w1 row in chunk
        const size_t brow1 = (size_t)(ic0 + ccol) * H;
        v8f acc1a = v8f_zero(), acc1b = v8f_zero();
        v16bf bcur = load_bfrag<BF16W>(w1e, brow1 + bkoff);
#pragma unroll
        for (int kk = 0; kk < H; kk += 32) {
            v16bf bnxt = bcur;
            if (kk + 32 < H)
                bnxt = load_bfrag<BF16W>(w1e, brow1 + kk + 32 + bkoff);
            v16bf a0 = load_afrag(&x_lds[ln16 * XSTR + kk + akoff]);
            v16bf a1 = load_afrag(&x_lds[(16 + ln16) * XSTR + kk + akoff]);
            acc1a = __builtin_amdgcn_wmma_f32_16x16x32_bf16(
                false, a0, false, bcur, (short)0, acc1a, false, false);
            acc1b = __builtin_amdgcn_wmma_f32_16x16x32_bf16(
                false, a1, false, bcur, (short)0, acc1b, false, false);
            bcur = bnxt;
        }

        __syncthreads();   // previous chunk's h_lds readers are done
#pragma unroll
        for (int r = 0; r < 8; ++r) {
            int m = r + hi * 8;
            h_lds[m * HSTR + wave * 16 + ln16]        = (bf16)gelu_tanh(acc1a[r]);
            h_lds[(16 + m) * HSTR + wave * 16 + ln16] = (bf16)gelu_tanh(acc1b[r]);
        }
        __syncthreads();   // h chunk ready

        // ---- GEMM2: y[32,768] += h[32,128] @ W2[:,chunk]^T ----
        // Issue ALL 6 B-fragments (12 x global_load_b128) per K-step before
        // the 12 consuming WMMAs: waits become staged instead of loadcnt==0.
#pragma unroll
        for (int kc = 0; kc < 128; kc += 32) {
            v16bf a0 = load_afrag(&h_lds[ln16 * HSTR + kc + akoff]);
            v16bf a1 = load_afrag(&h_lds[(16 + ln16) * HSTR + kc + akoff]);
            v16bf b[6];
#pragma unroll
            for (int j = 0; j < 6; ++j) {
                const int nb = (wave + 8 * j) * 16;           // output col base
                b[j] = load_bfrag<BF16W>(w2e, (size_t)(nb + ln16) * I + ic0 + kc + bkoff);
            }
#pragma unroll
            for (int j = 0; j < 6; ++j) {
                accY[2 * j]     = __builtin_amdgcn_wmma_f32_16x16x32_bf16(
                    false, a0, false, b[j], (short)0, accY[2 * j], false, false);
                accY[2 * j + 1] = __builtin_amdgcn_wmma_f32_16x16x32_bf16(
                    false, a1, false, b[j], (short)0, accY[2 * j + 1], false, false);
            }
        }
    }

    // ---- scale by combine weight, accumulate into y (2 experts per token) ----
#pragma unroll
    for (int j = 0; j < 6; ++j) {
        const int nb = (wave + 8 * j) * 16;
#pragma unroll
        for (int mt = 0; mt < 2; ++mt) {
#pragma unroll
            for (int r = 0; r < 8; ++r) {
                int m = 16 * mt + r + hi * 8;                 // C/D layout row
                int tok = trow[m];
                if (tok >= 0) {
                    float v = accY[2 * j + mt][r] * wrow[m];
                    atomicAdd(&y[(size_t)tok * H + nb + ln16], v);
                }
            }
        }
    }
}

// ---------------------------------------------------------------------------
extern "C" void kernel_launch(void* const* d_in, const int* in_sizes, int n_in,
                              void* d_out, int out_size, void* d_ws, size_t ws_size,
                              hipStream_t stream) {
    const float* x  = (const float*)d_in[0];   // [4,4096,768]
    const float* gw = (const float*)d_in[1];   // [8,768]
    const float* w1 = (const float*)d_in[2];   // [8,3072,768]
    const float* w2 = (const float*)d_in[3];   // [8,768,3072]
    float* y = (float*)d_out;                  // [N*H] y, then [1] aux_loss

    // ws layout: counts(32B)|psum(32B)|tok_idx|tok_w|w1_bf16|w2_bf16
    char*  ws      = (char*)d_ws;
    int*   counts  = (int*)ws;
    float* psum    = (float*)(ws + 32);
    int*   tok_idx = (int*)(ws + 64);
    float* tok_w   = (float*)(ws + 64 + (size_t)E * NTOK * sizeof(int));
    size_t lists_end = 64 + (size_t)E * NTOK * 8;             // 1 MB + 64 B
    bf16*  w1bf = (bf16*)(ws + lists_end);
    bf16*  w2bf = (bf16*)(ws + lists_end + (size_t)E * I * H * sizeof(bf16));
    size_t need = lists_end + 2 * (size_t)E * I * H * sizeof(bf16);  // ~76.5 MB

    const bool preconv = (ws_size >= need);

    k_zero  <<<4096, 256, 0, stream>>>(y, NTOK * H + 1, counts, psum);
    k_router<<<NTOK / 8, 256, 0, stream>>>(x, gw, counts, psum, tok_idx, tok_w);
    k_aux   <<<1, 32, 0, stream>>>(counts, psum, y + (size_t)NTOK * H);

    if (preconv) {
        k_cvtw<<<4096, 256, 0, stream>>>(w1, w1bf, E * I * H / 4);
        k_cvtw<<<4096, 256, 0, stream>>>(w2, w2bf, E * H * I / 4);
        k_ffn<true><<<E * TILES_MAX, 256, 0, stream>>>(
            x, w1bf, w2bf, counts, tok_idx, tok_w, y);
    } else {
        k_ffn<false><<<E * TILES_MAX, 256, 0, stream>>>(
            x, w1, w2, counts, tok_idx, tok_w, y);
    }
}
